// GCN_Module_18004502905149
// MI455X (gfx1250) — compile-verified, matrix operand order checked by reading, not run on hardware
//
#include <hip/hip_runtime.h>
#include <math.h>

typedef unsigned short u16;
typedef __attribute__((ext_vector_type(16))) __bf16 v16bf;
typedef __attribute__((ext_vector_type(8)))  float  v8f;

#define WMMA_BF16(A, B, C) \
  __builtin_amdgcn_wmma_f32_16x16x32_bf16(false, (A), false, (B), (short)0, (C), false, false)

// ---------- helpers ----------
__device__ __forceinline__ u16 f2bf(float x) {
  union { float f; unsigned u; } v; v.f = x;
  unsigned r = v.u + 0x7FFFu + ((v.u >> 16) & 1u);   // round-to-nearest-even
  return (u16)(r >> 16);
}
__device__ __forceinline__ float bf2f(u16 h) {
  union { unsigned u; float f; } v; v.u = ((unsigned)h) << 16; return v.f;
}
// load 16 bf16 as two 16B chunks (p0 -> elems 0..7, p1 -> elems 8..15)
__device__ __forceinline__ v16bf ldfrag(const u16* p0, const u16* p1) {
  v16bf f;
  reinterpret_cast<uint4*>(&f)[0] = *reinterpret_cast<const uint4*>(p0);
  reinterpret_cast<uint4*>(&f)[1] = *reinterpret_cast<const uint4*>(p1);
  return f;
}

// ---------- constants ----------
#define Bsz 512
#define Nn  12
#define NFG 1024
#define NFR 256
#define NG  16
#define MROWS (Bsz * Nn)               // 6144
#define OUT_OFF ((size_t)MROWS * NFG)  // rel tail offset in d_out

// ---------- kernel 1: fp32 -> bf16 ----------
__global__ void cvt_kernel(const float* __restrict__ src, u16* __restrict__ dst, int n) {
  int i = blockIdx.x * 256 + threadIdx.x;
  if (i < n) dst[i] = f2bf(src[i]);
}

// ---------- kernel 2: distance mask (0 or -inf) ----------
__global__ void mask_kernel(const float* __restrict__ box, float* __restrict__ mb) {
  int b = blockIdx.x;
  int t = threadIdx.x;                // 144 threads: (n,m)
  int n = t / Nn, m = t % Nn;
  const float* bi = box + (size_t)(b * Nn + n) * 4;
  const float* bj = box + (size_t)(b * Nn + m) * 4;
  float cxi = (bi[0] + bi[2]) * 0.5f, cyi = (bi[1] + bi[3]) * 0.5f;
  float cxj = (bj[0] + bj[2]) * 0.5f, cyj = (bj[1] + bj[3]) * 0.5f;
  float dx = cxi - cxj, dy = cyi - cyj;
  float d = sqrtf(fmaxf(dx * dx + dy * dy, 0.0f));
  mb[(size_t)b * 144 + t] = (d > 0.2f) ? -__builtin_inff() : 0.0f;
}

// ---------- kernel 3: theta/phi (WMMA bf16) + softmax -> rel ----------
// grid = B*G blocks, 512 threads (16 waves); wave w owns theta tile w and phi tile w
__global__ void attn_kernel(const u16* __restrict__ xbf,
                            const u16* __restrict__ wth, const u16* __restrict__ wph,
                            const float* __restrict__ bth, const float* __restrict__ bph,
                            const float* __restrict__ mb,
                            float* __restrict__ relws, float* __restrict__ rel_out) {
  // x staging (bf16, zero-padded rows 12..15) overlaid with theta/phi fp32.
  __shared__ union {
    __attribute__((aligned(16))) u16 x[16][NFG];                  // 32 KB (phase 1)
    struct { float th[16][NFR]; float ph[16][NFR]; } tp;          // 32 KB (phase 2)
  } sh;
  __shared__ float simsh[144];

  int bg = blockIdx.x;
  int g = bg & 15, b = bg >> 4;
  int tid = threadIdx.x;
  int w = tid >> 5, l = tid & 31;
  int m  = l & 15;
  int hi = (l & 16) ? 8 : 0;

  // stage x rows (pad 12..15 with zeros)
  for (int i = tid; i < 16 * NFG; i += 512) {
    int r = i >> 10, f = i & (NFG - 1);
    sh.x[r][f] = (r < Nn) ? xbf[((size_t)(b * Nn + r) << 10) + f] : (u16)0;
  }
  __syncthreads();

  v8f accT = (v8f){0.f,0.f,0.f,0.f,0.f,0.f,0.f,0.f};
  v8f accP = (v8f){0.f,0.f,0.f,0.f,0.f,0.f,0.f,0.f};

  // scalar bases + 32-bit per-lane element offsets (SADDR-form loads)
  const u16* wtbase = wth + (((size_t)g * NFR) << 10);
  const u16* wpbase = wph + (((size_t)g * NFR) << 10);
  int c = (w * 16 + m) << 10;            // column offset (elements)
  const u16* ap = &sh.x[m][0];

  auto ldA = [&](int kb) { return ldfrag(ap + kb + hi, ap + kb + 16 + hi); };
  auto ldB = [&](const u16* base, int kb) {
    return ldfrag(base + c + kb + hi, base + c + kb + 16 + hi);
  };

  // ping-pong pipelined K loop (no inter-buffer copies)
  v16bf a0 = ldA(0), a1;
  v16bf t0 = ldB(wtbase, 0), p0 = ldB(wpbase, 0);
  v16bf t1, p1;
  for (int kt = 0; kt < 30; kt += 2) {
    int k1 = (kt + 1) * 32, k2 = (kt + 2) * 32;
    a1 = ldA(k1); t1 = ldB(wtbase, k1); p1 = ldB(wpbase, k1);
    accT = WMMA_BF16(a0, t0, accT);
    accP = WMMA_BF16(a0, p0, accP);
    a0 = ldA(k2); t0 = ldB(wtbase, k2); p0 = ldB(wpbase, k2);
    accT = WMMA_BF16(a1, t1, accT);
    accP = WMMA_BF16(a1, p1, accP);
  }
  {
    int k1 = 31 * 32;
    a1 = ldA(k1); t1 = ldB(wtbase, k1); p1 = ldB(wpbase, k1);
    accT = WMMA_BF16(a0, t0, accT);
    accP = WMMA_BF16(a0, p0, accP);
    accT = WMMA_BF16(a1, t1, accT);
    accP = WMMA_BF16(a1, p1, accP);
  }
  __syncthreads();   // x phase done; reuse LDS for theta/phi

  // C layout: VGPR v -> row v (lanes 0-15) / v+8 (lanes 16-31); lane&15 -> column
  {
    int n = w * 16 + (l & 15);
    float btv = bth[g * NFR + n];
    float bpv = bph[g * NFR + n];
#pragma unroll
    for (int v = 0; v < 8; ++v) {
      sh.tp.th[v + hi][n] = accT[v] + btv;
      sh.tp.ph[v + hi][n] = accP[v] + bpv;
    }
  }
  __syncthreads();

  // sim = theta . phi^T / sqrt(256) + mask   (144 dots of length 256)
  if (tid < 144) {
    int n = tid / Nn, mm = tid % Nn;
    float dot = 0.0f;
    for (int k = 0; k < NFR; ++k) dot += sh.tp.th[n][k] * sh.tp.ph[mm][k];
    simsh[tid] = dot * 0.0625f + mb[(size_t)b * 144 + tid];
  }
  __syncthreads();

  if (tid < Nn) {
    int n = tid;
    float mx = -__builtin_inff();
    for (int mm = 0; mm < Nn; ++mm) mx = fmaxf(mx, simsh[n * Nn + mm]);
    float e[Nn]; float sum = 0.0f;
    for (int mm = 0; mm < Nn; ++mm) { e[mm] = __expf(simsh[n * Nn + mm] - mx); sum += e[mm]; }
    float inv = 1.0f / sum;
    float* dst = relws + ((size_t)(b * NG + g) * 144) + n * Nn;
    for (int mm = 0; mm < Nn; ++mm) {
      float r = e[mm] * inv;
      dst[mm] = r;
      if (g == NG - 1) rel_out[(size_t)(b * Nn + n) * Nn + mm] = r;
    }
  }
}

// ---------- kernel 4: agg -> h (WMMA bf16) -> LN -> relu -> sum over heads ----------
// grid = MROWS/16 = 384 blocks, 1024 threads (32 waves); wave w owns cols [w*32, w*32+32)
__global__ void gcn_kernel(const u16* __restrict__ xbf, const u16* __restrict__ wgcn,
                           const float* __restrict__ relws,
                           const float* __restrict__ gamma, const float* __restrict__ beta,
                           float* __restrict__ out) {
  __shared__ __attribute__((aligned(16))) u16 aggsh[16][NFG];   // 32 KB
  __shared__ float relsh[288];                                  // rel for 2 batches
  __shared__ float psum[16][32];                                // [row][wave]
  __shared__ float psq[16][32];
  __shared__ float rmu[16], rvar[16];

  int M0 = blockIdx.x * 16;
  int b0 = M0 / Nn;
  int tid = threadIdx.x;
  int w = tid >> 5, l = tid & 31;
  int m  = l & 15;
  int hi = (l & 16) ? 8 : 0;
  int ncol0 = w * 32;

  v8f oacc0 = (v8f){0.f,0.f,0.f,0.f,0.f,0.f,0.f,0.f};
  v8f oacc1 = (v8f){0.f,0.f,0.f,0.f,0.f,0.f,0.f,0.f};

  for (int g = 0; g < NG; ++g) {
    __syncthreads();   // protect LDS reuse from previous head
    if (tid < 288) {
      int bb = b0 + tid / 144;
      relsh[tid] = (bb < Bsz) ? relws[((size_t)(bb * NG + g) * 144) + (tid % 144)] : 0.0f;
    }
    __syncthreads();

    // agg[r][f] = sum_m rel[b,g,n,m] * x[b,m,f]  (bf16 into LDS)
    for (int i = tid; i < 16 * NFG; i += 1024) {
      int r = i >> 10, f = i & (NFG - 1);
      int row = M0 + r;
      int bb = row / Nn;
      int n  = row - bb * Nn;
      const float* rr = relsh + (bb - b0) * 144 + n * Nn;
      const u16* xp = xbf + (((size_t)bb * Nn) << 10) + f;
      float s = 0.0f;
#pragma unroll
      for (int mm = 0; mm < Nn; ++mm) s += rr[mm] * bf2f(xp[(size_t)mm << 10]);
      aggsh[r][f] = f2bf(s);
    }
    __syncthreads();

    // h tile: [16 rows] x [32 cols per wave], K = 1024 (ping-pong pipelined)
    const u16* wgg = wgcn + ((size_t)g << 20);
    int coff0 = (ncol0 + (l & 15)) << 10;
    int coff1 = coff0 + (16 << 10);
    const u16* ap = &aggsh[m][0];

    auto ldA = [&](int kb) { return ldfrag(ap + kb + hi, ap + kb + 16 + hi); };
    auto ldB = [&](int coffv, int kb) {
      return ldfrag(wgg + coffv + kb + hi, wgg + coffv + kb + 16 + hi);
    };

    v8f hc0 = (v8f){0.f,0.f,0.f,0.f,0.f,0.f,0.f,0.f};
    v8f hc1 = (v8f){0.f,0.f,0.f,0.f,0.f,0.f,0.f,0.f};

    v16bf a0 = ldA(0), a1;
    v16bf b0a = ldB(coff0, 0), b0b = ldB(coff1, 0);
    v16bf b1a, b1b;
    for (int kt = 0; kt < 30; kt += 2) {
      int k1 = (kt + 1) * 32, k2 = (kt + 2) * 32;
      a1 = ldA(k1);
      b1a = ldB(coff0, k1); b1b = ldB(coff1, k1);
      __builtin_prefetch(wgg + coff0 + k1 + 96, 0, 3);   // -> global_prefetch_b8
      __builtin_prefetch(wgg + coff1 + k1 + 96, 0, 3);
      hc0 = WMMA_BF16(a0, b0a, hc0);
      hc1 = WMMA_BF16(a0, b0b, hc1);
      a0 = ldA(k2);
      b0a = ldB(coff0, k2); b0b = ldB(coff1, k2);
      hc0 = WMMA_BF16(a1, b1a, hc0);
      hc1 = WMMA_BF16(a1, b1b, hc1);
    }
    {
      int k1 = 31 * 32;
      a1 = ldA(k1);
      b1a = ldB(coff0, k1); b1b = ldB(coff1, k1);
      hc0 = WMMA_BF16(a0, b0a, hc0);
      hc1 = WMMA_BF16(a0, b0b, hc1);
      hc0 = WMMA_BF16(a1, b1a, hc0);
      hc1 = WMMA_BF16(a1, b1b, hc1);
    }

    // deterministic LN reduction: fixed-tree shuffle within 16-lane row groups
#pragma unroll
    for (int v = 0; v < 8; ++v) {
      float x0 = hc0[v], x1 = hc1[v];
      float s  = x0 + x1;
      float s2 = x0 * x0 + x1 * x1;
#pragma unroll
      for (int off = 8; off > 0; off >>= 1) {
        s  += __shfl_xor(s,  off, 16);
        s2 += __shfl_xor(s2, off, 16);
      }
      if ((l & 15) == 0) { psum[v + hi][w] = s; psq[v + hi][w] = s2; }
    }
    __syncthreads();
    if (tid < 16) {
      float s = 0.0f, s2 = 0.0f;
      for (int i = 0; i < 32; ++i) { s += psum[tid][i]; s2 += psq[tid][i]; }
      float mu = s * (1.0f / NFG);
      rmu[tid]  = mu;
      rvar[tid] = s2 * (1.0f / NFG) - mu * mu;
    }
    __syncthreads();

    // LN + relu + accumulate over heads
    {
      int o0 = ncol0 + (l & 15);
      int o1 = o0 + 16;
      float gl0 = gamma[g * NFG + o0], bl0 = beta[g * NFG + o0];
      float gl1 = gamma[g * NFG + o1], bl1 = beta[g * NFG + o1];
#pragma unroll
      for (int v = 0; v < 8; ++v) {
        int row = v + hi;
        float rs = rsqrtf(rvar[row] + 1e-5f);
        float mu = rmu[row];
        float hn0 = (hc0[v] - mu) * rs * gl0 + bl0;
        float hn1 = (hc1[v] - mu) * rs * gl1 + bl1;
        oacc0[v] += fmaxf(hn0, 0.0f);
        oacc1[v] += fmaxf(hn1, 0.0f);
      }
    }
  }

  // store out [MROWS, NFG]
  {
    int o0 = ncol0 + (l & 15);
    int o1 = o0 + 16;
#pragma unroll
    for (int v = 0; v < 8; ++v) {
      out[(size_t)(M0 + v + hi) * NFG + o0] = oacc0[v];
      out[(size_t)(M0 + v + hi) * NFG + o1] = oacc1[v];
    }
  }
}

// ---------- host ----------
extern "C" void kernel_launch(void* const* d_in, const int* in_sizes, int n_in,
                              void* d_out, int out_size, void* d_ws, size_t ws_size,
                              hipStream_t stream) {
  (void)in_sizes; (void)n_in; (void)out_size; (void)ws_size;
  const float* x     = (const float*)d_in[0];   // [B,N,NFG]
  const float* box   = (const float*)d_in[1];   // [B*N,4]
  const float* Wth   = (const float*)d_in[2];   // [G,NFR,NFG]
  const float* bth   = (const float*)d_in[3];   // [G,NFR]
  const float* Wph   = (const float*)d_in[4];
  const float* bph   = (const float*)d_in[5];
  const float* Wgc   = (const float*)d_in[6];   // [G,NFG,NFG]
  const float* gam   = (const float*)d_in[7];
  const float* bet   = (const float*)d_in[8];

  float* out     = (float*)d_out;               // [B,N,NFG]
  float* rel_out = out + OUT_OFF;               // [B,N,N]

  char* ws = (char*)d_ws;
  size_t off = 0;
  u16* xbf  = (u16*)(ws + off); off += (size_t)MROWS * NFG * 2;        // 12.6 MB
  u16* wthb = (u16*)(ws + off); off += (size_t)NG * NFR * NFG * 2;     //  8.4 MB
  u16* wphb = (u16*)(ws + off); off += (size_t)NG * NFR * NFG * 2;     //  8.4 MB
  u16* wgcb = (u16*)(ws + off); off += (size_t)NG * NFG * NFG * 2;     // 33.6 MB
  float* mb    = (float*)(ws + off); off += (size_t)Bsz * 144 * 4;     //  0.3 MB
  float* relws = (float*)(ws + off); off += (size_t)Bsz * NG * 144 * 4;//  4.7 MB

  // fp32 -> bf16 conversions
  int nx = MROWS * NFG;               // 6,291,456
  int nw = NG * NFR * NFG;            // 4,194,304
  int ng = NG * NFG * NFG;            // 16,777,216
  cvt_kernel<<<(nx + 255) / 256, 256, 0, stream>>>(x,   xbf,  nx);
  cvt_kernel<<<(nw + 255) / 256, 256, 0, stream>>>(Wth, wthb, nw);
  cvt_kernel<<<(nw + 255) / 256, 256, 0, stream>>>(Wph, wphb, nw);
  cvt_kernel<<<(ng + 255) / 256, 256, 0, stream>>>(Wgc, wgcb, ng);

  mask_kernel<<<Bsz, 144, 0, stream>>>(box, mb);

  attn_kernel<<<Bsz * NG, 512, 0, stream>>>(xbf, wthb, wphb, bth, bph, mb,
                                            relws, rel_out);

  gcn_kernel<<<MROWS / 16, 1024, 0, stream>>>(xbf, wgcb, relws, gam, bet, out);
}
